// HPUWeightOnlyLinear_86071144611917
// MI455X (gfx1250) — matrix operand order repaired and checked by reading, not compiled, not run on hardware
//
#include <hip/hip_runtime.h>

// Problem constants (from the reference)
#define IN_F   4096
#define OUT_F  11008
#define GROUPK 128
#define NPACK  (OUT_F / 8)   // 1376 packed int32 per K row

// Tiling
#define BM 128
#define BN 128
#define BK 32
#define PAD 8
#define LDA (BK + PAD)       // 40 bf16 elems -> 80B row stride (16B aligned)
#define LDB (BK + PAD)

typedef __bf16 bf16x2 __attribute__((ext_vector_type(2)));
typedef __bf16 bf16x4 __attribute__((ext_vector_type(4)));
typedef __bf16 bf16x8 __attribute__((ext_vector_type(8)));
typedef __bf16 v16bf  __attribute__((ext_vector_type(16)));
typedef float  v8f    __attribute__((ext_vector_type(8)));

union Frag { v16bf v; bf16x8 h[2]; };

__global__ __launch_bounds__(256, 2)
void woq_int4_bf16_wmma_kernel(const float* __restrict__ A,
                               const int*   __restrict__ qweight,
                               const int*   __restrict__ qzeros,
                               const float* __restrict__ scales,
                               float* __restrict__ C,
                               int M)
{
    __shared__ __align__(16) __bf16 As[2][BM][LDA];
    __shared__ __align__(16) __bf16 Bt[2][BN][LDB];   // transposed: [n][k]

    const int tid = threadIdx.x;
    const int bn0 = blockIdx.x * BN;
    const int bm0 = blockIdx.y * BM;

    const int lane  = tid & 31;     // wave32
    const int wave  = tid >> 5;     // 0..7
    const int wm    = wave >> 2;    // 0..1  -> 64 rows each
    const int wn    = wave & 3;     // 0..3  -> 32 cols each
    const int lrow  = lane & 15;
    const int lhalf = lane >> 4;

    // Weight-staging ownership: one packed column + a pair of adjacent k rows.
    const int pc  = tid & 15;           // packed col 0..15
    const int kr0 = (tid >> 4) * 2;     // even k row in tile: 0,2,...,30
    const int np  = (bn0 >> 3) + pc;    // global packed col
    const int n0  = pc * 8;             // col within tile

    // Per-group dequant constants held in registers: d = w*sc + (-zp*sc)
    float sc8[8], bi8[8];
    auto load_group = [&](int g) {
        const unsigned qz = (unsigned)qzeros[(size_t)g * NPACK + np];
        const float4 s0 = *(const float4*)(scales + (size_t)g * OUT_F + bn0 + n0);
        const float4 s1 = *(const float4*)(scales + (size_t)g * OUT_F + bn0 + n0 + 4);
        sc8[0] = s0.x; sc8[1] = s0.y; sc8[2] = s0.z; sc8[3] = s0.w;
        sc8[4] = s1.x; sc8[5] = s1.y; sc8[6] = s1.z; sc8[7] = s1.w;
        #pragma unroll
        for (int j = 0; j < 8; ++j) {
            const float zp = (float)((qz >> (4 * j)) & 15u);
            bi8[j] = -zp * sc8[j];
        }
    };

    // ---- stage one K-tile (global -> LDS, with dequant for weights) ----
    auto load_tile = [&](int kt, int stage) {
        const int k0 = kt * BK;
        // A tile: 128 rows x 32 fp32 cols -> bf16. 256 thr * 4 float4 each.
        {
            const int ar = tid >> 3;          // 0..31
            const int ac = (tid & 7) * 4;     // 0,4,...,28
            #pragma unroll
            for (int i = 0; i < 4; ++i) {
                const int row = ar + i * 32;  // 0..127
                const float4 v =
                    *(const float4*)(A + (size_t)(bm0 + row) * IN_F + k0 + ac);
                bf16x4 pv;
                pv.x = (__bf16)v.x; pv.y = (__bf16)v.y;
                pv.z = (__bf16)v.z; pv.w = (__bf16)v.w;
                *(bf16x4*)(&As[stage][row][ac]) = pv;   // ds_store_b64
            }
        }
        // Weight tile: this thread covers k rows kr0, kr0+1 of packed col pc.
        {
            const int kg = k0 + kr0;
            const unsigned qw0 = (unsigned)qweight[(size_t)kg * NPACK + np];
            const unsigned qw1 = (unsigned)qweight[(size_t)(kg + 1) * NPACK + np];
            #pragma unroll
            for (int j = 0; j < 8; ++j) {
                const float d0 = fmaf((float)((qw0 >> (4 * j)) & 15u), sc8[j], bi8[j]);
                const float d1 = fmaf((float)((qw1 >> (4 * j)) & 15u), sc8[j], bi8[j]);
                bf16x2 p;
                p.x = (__bf16)d0;   // k = kr0
                p.y = (__bf16)d1;   // k = kr0+1
                *(bf16x2*)(&Bt[stage][n0 + j][kr0]) = p;  // ds_store_b32
            }
        }
    };

    // ---- accumulators ----
    v8f acc[4][2];
    const v8f vzero = {0.f, 0.f, 0.f, 0.f, 0.f, 0.f, 0.f, 0.f};
    #pragma unroll
    for (int mi = 0; mi < 4; ++mi)
        #pragma unroll
        for (int ni = 0; ni < 2; ++ni)
            acc[mi][ni] = vzero;

    // ---- one K-step of WMMA math from LDS ----
    auto compute = [&](int stage) {
        Frag a[4], b[2];
        #pragma unroll
        for (int mi = 0; mi < 4; ++mi) {
            // 16-bit A 16x32 layout: lane half selects K-octet
            const __bf16* p = &As[stage][wm * 64 + mi * 16 + lrow][0];
            a[mi].h[0] = *(const bf16x8*)(p + lhalf * 8);
            a[mi].h[1] = *(const bf16x8*)(p + 16 + lhalf * 8);
        }
        #pragma unroll
        for (int ni = 0; ni < 2; ++ni) {
            // B 32x16 layout: lane half selects K-octet (same as A)
            const __bf16* p = &Bt[stage][wn * 32 + ni * 16 + lrow][0];
            b[ni].h[0] = *(const bf16x8*)(p + lhalf * 8);
            b[ni].h[1] = *(const bf16x8*)(p + 16 + lhalf * 8);
        }
        #pragma unroll
        for (int mi = 0; mi < 4; ++mi)
            #pragma unroll
            for (int ni = 0; ni < 2; ++ni)
                acc[mi][ni] = __builtin_amdgcn_wmma_f32_16x16x32_bf16(
                    false, a[mi].v, false, b[ni].v,
                    (short)0, acc[mi][ni], false, false);
    };

    // ---- double-buffered main loop over K ----
    const int NT = IN_F / BK;   // 128 tiles, 4 per quant group
    load_group(0);
    load_tile(0, 0);
    __syncthreads();
    int stage = 0;
    for (int t = 0; t < NT; ++t) {
        const int nxt = stage ^ 1;
        if (t + 2 < NT) {
            // Keep L2 ahead of the pipeline (emits global_prefetch_b8)
            __builtin_prefetch(A + (size_t)(bm0 + (tid >> 3)) * IN_F + (t + 2) * BK, 0, 1);
            __builtin_prefetch(qweight + (size_t)((t + 2) * BK + kr0) * NPACK + np, 0, 1);
        }
        if (t + 1 < NT) {
            if (((t + 1) & 3) == 0) load_group((t + 1) >> 2);  // new quant group
            load_tile(t + 1, nxt);
        }
        compute(stage);
        __syncthreads();
        stage = nxt;
    }

    // ---- epilogue: C/D layout -> global fp32 stores ----
    #pragma unroll
    for (int mi = 0; mi < 4; ++mi)
        #pragma unroll
        for (int ni = 0; ni < 2; ++ni) {
            const int gm = bm0 + wm * 64 + mi * 16 + lhalf * 8;
            const int gn = bn0 + wn * 32 + ni * 16 + lrow;
            float* cp = C + (size_t)gm * OUT_F + gn;
            #pragma unroll
            for (int v = 0; v < 8; ++v)
                cp[(size_t)v * OUT_F] = acc[mi][ni][v];
        }
    (void)M;
}

extern "C" void kernel_launch(void* const* d_in, const int* in_sizes, int n_in,
                              void* d_out, int out_size, void* d_ws, size_t ws_size,
                              hipStream_t stream) {
    const float* A       = (const float*)d_in[0];  // (2,2048,4096) fp32
    const int*   qweight = (const int*)  d_in[1];  // (4096,1376) int32
    const int*   qzeros  = (const int*)  d_in[2];  // (32,1376) int32
    const float* scales  = (const float*)d_in[3];  // (32,11008) fp32
    float*       C       = (float*)d_out;          // (2,2048,11008) fp32

    const int M = in_sizes[0] / IN_F;              // 4096
    dim3 grid(OUT_F / BN, M / BM);                 // (86, 32)
    woq_int4_bf16_wmma_kernel<<<grid, dim3(256), 0, stream>>>(
        A, qweight, qzeros, scales, C, M);
    (void)n_in; (void)out_size; (void)d_ws; (void)ws_size;
}
